// Block_39599598469865
// MI455X (gfx1250) — compile-verified, matrix operand order checked by reading, not run on hardware
//
#include <hip/hip_runtime.h>
#include <hip/hip_bf16.h>

typedef __attribute__((ext_vector_type(16))) __bf16 v16bf;
typedef __attribute__((ext_vector_type(8)))  float  v8f;

#define Bq    4
#define Nq    4096
#define Cq    768
#define HEADS 12
#define PTS   4
#define HD    64
#define HID   3072
#define M_TOT (Bq*Nq)   // 16384

// ---------------------------------------------------------------- fp32->bf16
__global__ void k_cvt_bf16(const float* __restrict__ s, __bf16* __restrict__ d, int n) {
  int i = blockIdx.x * blockDim.x + threadIdx.x;
  if (i < n) d[i] = (__bf16)s[i];
}

// ------------------------------------------------- LayerNorm (C=768) -> bf16
__global__ __launch_bounds__(256)
void k_layernorm_bf16(const float* __restrict__ x, const float* __restrict__ w,
                      const float* __restrict__ b, __bf16* __restrict__ out) {
  __shared__ float red[256];
  const int row = blockIdx.x;
  const int t = threadIdx.x;
  const float* xr = x + (size_t)row * Cq;
  float v0 = xr[t], v1 = xr[t + 256], v2 = xr[t + 512];
  red[t] = v0 + v1 + v2;
  __syncthreads();
  for (int o = 128; o; o >>= 1) { if (t < o) red[t] += red[t + o]; __syncthreads(); }
  float mean = red[0] * (1.0f / Cq);
  __syncthreads();
  float d0 = v0 - mean, d1 = v1 - mean, d2 = v2 - mean;
  red[t] = d0 * d0 + d1 * d1 + d2 * d2;
  __syncthreads();
  for (int o = 128; o; o >>= 1) { if (t < o) red[t] += red[t + o]; __syncthreads(); }
  float rstd = rsqrtf(red[0] * (1.0f / Cq) + 1e-5f);
  __bf16* orow = out + (size_t)row * Cq;
  orow[t]       = (__bf16)(d0 * rstd * w[t]       + b[t]);
  orow[t + 256] = (__bf16)(d1 * rstd * w[t + 256] + b[t + 256]);
  orow[t + 512] = (__bf16)(d2 * rstd * w[t + 512] + b[t + 512]);
}

// --------------------------------------------------------- bf16 WMMA GEMM
// out[m,n] = sum_k A[m,k] * W[n,k]  (+bias, opt GELU, opt residual)
// Block tile 128x64, 8 waves, each wave 32x32 via four 16x16x32 WMMAs.
// Double-buffered LDS fed by async global->LDS copies (ASYNCcnt).
#define BM 128
#define BN 64
#define BK 32
#define LDS_STRIDE 40   // 32 + 8 bf16 pad (80B rows, 16B aligned)
#define A_TILE (BM * LDS_STRIDE)   // bf16 elems per A buffer
#define B_TILE (BN * LDS_STRIDE)   // bf16 elems per B buffer

__global__ __launch_bounds__(256)
void k_gemm_bf16(const __bf16* __restrict__ A, int lda,
                 const __bf16* __restrict__ Wt,          // [Nout, K]
                 const float* __restrict__ bias,
                 const float* __restrict__ resid, int ldr,
                 float* __restrict__ outf, int ldo,
                 __bf16* __restrict__ outb, int ldob, int nbf,
                 int M, int Nout, int K, int gelu)
{
  __shared__ __bf16 As[2 * A_TILE];
  __shared__ __bf16 Bs[2 * B_TILE];
  const int tid  = threadIdx.x;
  const int lane = tid & 31, wid = tid >> 5;
  const int m0 = blockIdx.y * BM;
  const int n0 = blockIdx.x * BN;
  const int wm = (wid & 3) * 32;        // 4 waves cover 128 rows
  const int wn = (wid >> 2) * 32;       // 2 waves cover 64 cols
  const int lrow  = lane & 15;
  const int kbase = (lane < 16) ? 0 : 8;  // ISA 16-bit A/B lane K-pattern

  // LDS byte offsets of the staging buffers (generic ptr low 32 bits = LDS addr)
  const unsigned asBase = (unsigned)(uintptr_t)(void*)As;
  const unsigned bsBase = (unsigned)(uintptr_t)(void*)Bs;

  // this thread's staging assignments
  const int ar0 = tid >> 2,        ac0 = (tid & 3) * 8;         // A chunk 0
  const int ar1 = (tid + 256) >> 2, ac1 = ((tid + 256) & 3) * 8; // A chunk 1
  const int br  = tid >> 2,        bc  = (tid & 3) * 8;          // B chunk
  const bool bok = (n0 + br) < Nout;

  // stage tile k0 into buffer `buf` via async global->LDS copies
  auto stage = [&](int buf, int k0) {
    unsigned aL0 = asBase + (unsigned)(buf * A_TILE + ar0 * LDS_STRIDE + ac0) * 2u;
    unsigned aL1 = asBase + (unsigned)(buf * A_TILE + ar1 * LDS_STRIDE + ac1) * 2u;
    unsigned long long aG0 =
        (unsigned long long)(uintptr_t)(A + (size_t)(m0 + ar0) * lda + k0 + ac0);
    unsigned long long aG1 =
        (unsigned long long)(uintptr_t)(A + (size_t)(m0 + ar1) * lda + k0 + ac1);
    asm volatile("global_load_async_to_lds_b128 %0, %1, off"
                 :: "v"(aL0), "v"(aG0) : "memory");
    asm volatile("global_load_async_to_lds_b128 %0, %1, off"
                 :: "v"(aL1), "v"(aG1) : "memory");
    if (bok) {
      unsigned bL = bsBase + (unsigned)(buf * B_TILE + br * LDS_STRIDE + bc) * 2u;
      unsigned long long bG =
          (unsigned long long)(uintptr_t)(Wt + (size_t)(n0 + br) * K + k0 + bc);
      asm volatile("global_load_async_to_lds_b128 %0, %1, off"
                   :: "v"(bL), "v"(bG) : "memory");
    } else {
      uint4 z = {0u, 0u, 0u, 0u};
      *(uint4*)(Bs + buf * B_TILE + br * LDS_STRIDE + bc) = z;  // zero-fill OOB
    }
  };

  v8f acc[2][2] = {};
  const int nk = K / BK;

  stage(0, 0);
  asm volatile("s_wait_asynccnt 0x0" ::: "memory");
  __syncthreads();

  for (int t = 0; t < nk; ++t) {
    const int buf = t & 1;
    if (t + 1 < nk) stage(buf ^ 1, (t + 1) * BK);  // prefetch next tile (async)

    const __bf16* Ab = As + buf * A_TILE;
    const __bf16* Bb = Bs + buf * B_TILE;
    v16bf afr[2], bfr[2];
#pragma unroll
    for (int mi = 0; mi < 2; ++mi) {
      const __bf16* p = Ab + (wm + mi * 16 + lrow) * LDS_STRIDE + kbase;
      union { uint4 u[2]; v16bf v; } tu;
      tu.u[0] = *(const uint4*)p;
      tu.u[1] = *(const uint4*)(p + 16);
      afr[mi] = tu.v;
    }
#pragma unroll
    for (int ni = 0; ni < 2; ++ni) {
      const __bf16* p = Bb + (wn + ni * 16 + lrow) * LDS_STRIDE + kbase;
      union { uint4 u[2]; v16bf v; } tu;
      tu.u[0] = *(const uint4*)p;
      tu.u[1] = *(const uint4*)(p + 16);
      bfr[ni] = tu.v;
    }
#pragma unroll
    for (int mi = 0; mi < 2; ++mi)
#pragma unroll
      for (int ni = 0; ni < 2; ++ni)
        acc[mi][ni] = __builtin_amdgcn_wmma_f32_16x16x32_bf16(
            false, afr[mi], false, bfr[ni], (short)0, acc[mi][ni], false, false);

    // ensure next buffer's async copies landed; barrier also orders LDS reads
    asm volatile("s_wait_asynccnt 0x0" ::: "memory");
    __syncthreads();
  }

  // epilogue: C/D layout — VGPR r: lanes0-15 -> M=r, lanes16-31 -> M=8+r, N=lane%16
  const int rowAdd = (lane < 16) ? 0 : 8;
  const int colL = lane & 15;
#pragma unroll
  for (int mi = 0; mi < 2; ++mi) {
#pragma unroll
    for (int ni = 0; ni < 2; ++ni) {
      int col = n0 + wn + ni * 16 + colL;
      if (col >= Nout) continue;
      float bv = bias ? bias[col] : 0.0f;
#pragma unroll
      for (int r = 0; r < 8; ++r) {
        int row = m0 + wm + mi * 16 + rowAdd + r;
        float v = acc[mi][ni][r] + bv;
        if (gelu) v = 0.5f * v * (1.0f + erff(v * 0.70710678118f));
        if (resid) v += resid[(size_t)row * ldr + col];
        if (outf) outf[(size_t)row * ldo + col] = v;
        if (outb && col < nbf) outb[(size_t)row * ldob + col] = (__bf16)v;
      }
    }
  }
}

// --------------------------------- deformable sampling + attention + softmax
// one wave per (b, h, n); 2 channels per lane (HD=64 / 32 lanes)
__global__ __launch_bounds__(256)
void k_deform_attn(const float* __restrict__ qkv,   // [M_TOT, 3C]
                   const float* __restrict__ off,   // [M_TOT, HEADS*PTS*2]
                   const float* __restrict__ ref,   // [M_TOT, 2]
                   const int* __restrict__ Hp, const int* __restrict__ Wp,
                   __bf16* __restrict__ out)        // [M_TOT, C] bf16
{
  const int lane = threadIdx.x & 31;
  const int wv = blockIdx.x * (blockDim.x >> 5) + (threadIdx.x >> 5);
  const int h  = wv % HEADS;
  const int bn = wv / HEADS;          // b*Nq + n
  const int b  = bn / Nq;

  const int W = *Wp, H = *Hp;
  const float Wf = (float)W, Hf = (float)H;

  const float rx = ref[(size_t)bn * 2 + 0];
  const float ry = ref[(size_t)bn * 2 + 1];
  const size_t rowbase = (size_t)bn * (3 * Cq);
  const int c0 = h * HD + 2 * lane;
  const float q0 = qkv[rowbase + c0];
  const float q1 = qkv[rowbase + c0 + 1];

  float attn[PTS], vx[PTS], vy[PTS];
#pragma unroll
  for (int p = 0; p < PTS; ++p) {
    float ox = off[(size_t)bn * (HEADS * PTS * 2) + h * (PTS * 2) + p * 2 + 0];
    float oy = off[(size_t)bn * (HEADS * PTS * 2) + h * (PTS * 2) + p * 2 + 1];
    // x = loc_x*W - 0.5 (algebraically identical to reference grid mapping)
    float xs = (rx + ox / Wf) * Wf - 0.5f;
    float ys = (ry + oy / Hf) * Hf - 0.5f;
    float x0f = floorf(xs), y0f = floorf(ys);
    float wx = xs - x0f, wy = ys - y0f;
    int x0 = (int)x0f, y0 = (int)y0f;
    float ks0 = 0.f, ks1 = 0.f, vs0 = 0.f, vs1 = 0.f;
#pragma unroll
    for (int cy = 0; cy < 2; ++cy) {
#pragma unroll
      for (int cx = 0; cx < 2; ++cx) {
        int ix = x0 + cx, iy = y0 + cy;
        float wgt = (cx ? wx : 1.f - wx) * (cy ? wy : 1.f - wy);
        if ((ix >= 0) & (ix < W) & (iy >= 0) & (iy < H)) {
          size_t base = ((size_t)(b * Nq + iy * W + ix)) * (3 * Cq) + c0;
          float2 kv = *(const float2*)(qkv + base + Cq);       // k slice
          float2 vv = *(const float2*)(qkv + base + 2 * Cq);   // v slice
          ks0 += wgt * kv.x; ks1 += wgt * kv.y;
          vs0 += wgt * vv.x; vs1 += wgt * vv.y;
        }
      }
    }
    float d = q0 * ks0 + q1 * ks1;
#pragma unroll
    for (int o = 16; o; o >>= 1) d += __shfl_xor(d, o, 32);
    attn[p] = d * 0.125f;   // SCALE = HD^-0.5
    vx[p] = vs0; vy[p] = vs1;
  }
  float mx = fmaxf(fmaxf(attn[0], attn[1]), fmaxf(attn[2], attn[3]));
  float es = 0.f;
#pragma unroll
  for (int p = 0; p < PTS; ++p) { attn[p] = expf(attn[p] - mx); es += attn[p]; }
  float inv = 1.0f / es;
  float o0 = 0.f, o1 = 0.f;
#pragma unroll
  for (int p = 0; p < PTS; ++p) { float w = attn[p] * inv; o0 += w * vx[p]; o1 += w * vy[p]; }
  __bf16* orow = out + (size_t)bn * Cq + c0;
  orow[0] = (__bf16)o0;
  orow[1] = (__bf16)o1;
}

// --------------------------------------------------------------------- host
extern "C" void kernel_launch(void* const* d_in, const int* in_sizes, int n_in,
                              void* d_out, int out_size, void* d_ws, size_t ws_size,
                              hipStream_t stream) {
  const float* x      = (const float*)d_in[0];
  const float* ref    = (const float*)d_in[1];
  const float* n1w    = (const float*)d_in[2];
  const float* n1b    = (const float*)d_in[3];
  const float* qkv_w  = (const float*)d_in[4];
  const float* qkv_b  = (const float*)d_in[5];
  const float* off_w  = (const float*)d_in[6];
  const float* off_b  = (const float*)d_in[7];
  const float* proj_w = (const float*)d_in[8];
  const float* proj_b = (const float*)d_in[9];
  const float* n2w    = (const float*)d_in[10];
  const float* n2b    = (const float*)d_in[11];
  const float* fc1_w  = (const float*)d_in[12];
  const float* fc1_b  = (const float*)d_in[13];
  const float* fc2_w  = (const float*)d_in[14];
  const float* fc2_b  = (const float*)d_in[15];
  const int*   Hp     = (const int*)d_in[16];
  const int*   Wp     = (const int*)d_in[17];

  char* ws = (char*)d_ws;
  size_t p = 0;
  auto alloc = [&](size_t bytes) -> char* {
    char* r = ws + p;
    p += (bytes + 255) & ~(size_t)255;
    return r;
  };

  __bf16* qkvw_b = (__bf16*)alloc((size_t)2304 * 768 * 2);
  __bf16* offw_b = (__bf16*)alloc((size_t)96 * 768 * 2);
  __bf16* projw_b= (__bf16*)alloc((size_t)768 * 768 * 2);
  __bf16* fc1w_b = (__bf16*)alloc((size_t)3072 * 768 * 2);
  __bf16* fc2w_b = (__bf16*)alloc((size_t)768 * 3072 * 2);
  __bf16* h_b    = (__bf16*)alloc((size_t)M_TOT * 768 * 2);
  float*  qkv_f  = (float*) alloc((size_t)M_TOT * 2304 * 4);
  __bf16* q_b    = (__bf16*)alloc((size_t)M_TOT * 768 * 2);
  float*  off_f  = (float*) alloc((size_t)M_TOT * 96 * 4);
  __bf16* ao_b   = (__bf16*)alloc((size_t)M_TOT * 768 * 2);
  float*  x1_f   = (float*) alloc((size_t)M_TOT * 768 * 4);
  __bf16* h2_b   = (__bf16*)alloc((size_t)M_TOT * 768 * 2);
  __bf16* m_b    = (__bf16*)alloc((size_t)M_TOT * 3072 * 2);

  dim3 blk(256);
  // weight conversions (L2-resident, negligible)
  k_cvt_bf16<<<(2304*768 + 255)/256, blk, 0, stream>>>(qkv_w, qkvw_b, 2304*768);
  k_cvt_bf16<<<(96*768   + 255)/256, blk, 0, stream>>>(off_w, offw_b, 96*768);
  k_cvt_bf16<<<(768*768  + 255)/256, blk, 0, stream>>>(proj_w, projw_b, 768*768);
  k_cvt_bf16<<<(3072*768 + 255)/256, blk, 0, stream>>>(fc1_w, fc1w_b, 3072*768);
  k_cvt_bf16<<<(768*3072 + 255)/256, blk, 0, stream>>>(fc2_w, fc2w_b, 768*3072);

  // LN1
  k_layernorm_bf16<<<M_TOT, blk, 0, stream>>>(x, n1w, n1b, h_b);

  // QKV GEMM: fp32 out + bf16 mirror of q (first 768 cols)
  dim3 g1(2304/BN, M_TOT/BM);
  k_gemm_bf16<<<g1, blk, 0, stream>>>(h_b, 768, qkvw_b, qkv_b, nullptr, 0,
                                      qkv_f, 2304, q_b, 768, 768,
                                      M_TOT, 2304, 768, 0);
  // offsets GEMM (Nout=96)
  dim3 g2((96 + BN - 1)/BN, M_TOT/BM);
  k_gemm_bf16<<<g2, blk, 0, stream>>>(q_b, 768, offw_b, off_b, nullptr, 0,
                                      off_f, 96, nullptr, 0, 0,
                                      M_TOT, 96, 768, 0);
  // deformable sampling + attention
  k_deform_attn<<<(M_TOT*HEADS)/8, blk, 0, stream>>>(qkv_f, off_f, ref, Hp, Wp, ao_b);

  // proj GEMM + residual x
  dim3 g3(768/BN, M_TOT/BM);
  k_gemm_bf16<<<g3, blk, 0, stream>>>(ao_b, 768, projw_b, proj_b, x, 768,
                                      x1_f, 768, nullptr, 0, 0,
                                      M_TOT, 768, 768, 0);
  // LN2
  k_layernorm_bf16<<<M_TOT, blk, 0, stream>>>(x1_f, n2w, n2b, h2_b);

  // FC1 + GELU -> bf16
  dim3 g4(3072/BN, M_TOT/BM);
  k_gemm_bf16<<<g4, blk, 0, stream>>>(h2_b, 768, fc1w_b, fc1_b, nullptr, 0,
                                      nullptr, 0, m_b, 3072, 3072,
                                      M_TOT, 3072, 768, 1);
  // FC2 + residual -> d_out (fp32)
  dim3 g5(768/BN, M_TOT/BM);
  k_gemm_bf16<<<g5, blk, 0, stream>>>(m_b, 3072, fc2w_b, fc2_b, x1_f, 768,
                                      (float*)d_out, 768, nullptr, 0, 0,
                                      M_TOT, 768, 3072, 0);
}
